// LIF_20521353740818
// MI455X (gfx1250) — matrix operand order repaired and checked by reading, not compiled, not run on hardware
//
#include <hip/hip_runtime.h>

// LIF network: T sequential steps, each = GEMV(g, W[2048x2048]) + pointwise update.
// Persistent-grid design for MI455X (gfx1250, wave32):
//   32 WGs x 256 threads, each WG owns 64 output neurons, its 2048x64 f16 weight
//   slice lives in LDS (256 KB of the 320 KB/WGP). GEMV via v_wmma_f32_16x16x32_f16
//   (B operand fully utilized -> minimal WMMA count). One atomic grid barrier +
//   ping-pong f16 g broadcast (4 KB through L2) per step.
// Inner loop is branch-free (non-row-0 lanes read a zeroed LDS block) and uses
// 4 independent accumulators to break the WMMA D->C dependency chain.

#define TSTEPS 16384
#define NN 2048
#define NWG 32
#define NPW 64              // neurons per workgroup
#define THREADS 256
#define KBLK 32             // K per f16 WMMA
#define NKB (NN / KBLK)     // 64 k-blocks
#define FRAG_BYTES 1024     // one 32x16 f16 B fragment
#define WG_FRAG_BYTES (4 * NKB * FRAG_BYTES)   // 4 col-tiles * 64 * 1KB = 256 KB
#define GLDS_OFF   WG_FRAG_BYTES               // 4 KB g broadcast
#define ZBUF_OFF   (WG_FRAG_BYTES + 4096)      // 64 B zero block (A filler)
#define PART_OFF   (WG_FRAG_BYTES + 4096 + 64) // 2*64 floats partials
#define SMEM_BYTES (PART_OFF + 512)

typedef __attribute__((ext_vector_type(16))) _Float16 v16h;
typedef __attribute__((ext_vector_type(8)))  _Float16 v8h;
typedef __attribute__((ext_vector_type(8)))  float    v8f;
typedef __attribute__((ext_vector_type(4)))  float    v4f;

// ---- one-time weight prep: fp32 W (diag masked) -> f16 B-fragment layout in ws ----
// Fragment (kbg, col-tile ct) layout, contiguous K per lane half (A reads g in the same order):
//   lanes 0-15: column n = lane, K = kbg*32 + 0..15  (2 halves per VGPR, in order)
//   lanes 16-31: column n = lane-16, K = kbg*32 + 16..31
// Stored as [lane][halves 0..7] (16B) then at +512 [lane][halves 8..15].
__global__ void lif_prep(const float* __restrict__ w, _Float16* __restrict__ wf) {
    int tid  = blockIdx.x * blockDim.x + threadIdx.x;   // 262144 threads total
    int lane = tid & 31;
    int kbg  = (tid >> 5) & (NKB - 1);
    int ct   = (tid >> 11) & 3;
    int wg   = tid >> 13;
    int n    = wg * NPW + ct * 16 + (lane & 15);
    int kb0  = kbg * KBLK + ((lane >> 4) << 4);
    v8h lo, hi;
#pragma unroll
    for (int h = 0; h < 8; ++h) {
        int k0 = kb0 + h, k1 = kb0 + 8 + h;
        float a = (k0 == n) ? 0.f : w[(long)k0 * NN + n];   // w_eff: zero diagonal
        float b = (k1 == n) ? 0.f : w[(long)k1 * NN + n];
        lo[h] = (_Float16)a;
        hi[h] = (_Float16)b;
    }
    char* base = (char*)wf + (long)wg * WG_FRAG_BYTES + (ct * NKB + kbg) * FRAG_BYTES;
    *(v8h*)(base + lane * 16)       = lo;
    *(v8h*)(base + 512 + lane * 16) = hi;
}

__device__ __forceinline__ void grid_barrier(unsigned* bar, int tid, unsigned epoch) {
    __threadfence();
    __syncthreads();
    if (tid == 0) {
        unsigned arr = __hip_atomic_fetch_add(bar, 1u, __ATOMIC_ACQ_REL,
                                              __HIP_MEMORY_SCOPE_AGENT);
        if (arr == (unsigned)NWG * epoch - 1u)
            __hip_atomic_store(bar + 16, epoch, __ATOMIC_RELEASE,
                               __HIP_MEMORY_SCOPE_AGENT);
        while (__hip_atomic_load(bar + 16, __ATOMIC_ACQUIRE,
                                 __HIP_MEMORY_SCOPE_AGENT) < epoch)
            __builtin_amdgcn_s_sleep(1);
    }
    __syncthreads();
    __builtin_amdgcn_fence(__ATOMIC_ACQUIRE, "agent");
}

__global__ void __launch_bounds__(THREADS, 1) lif_persist(
    const float* __restrict__ x_in, const float* __restrict__ E_L,
    const float* __restrict__ tau_m, const float* __restrict__ R_I,
    const float* __restrict__ tau_g, const float* __restrict__ v0,
    const float* __restrict__ g0, float* __restrict__ out,
    unsigned* bar, _Float16* g_pub, const _Float16* __restrict__ wf_g)
{
    extern __shared__ char smem[];
    _Float16* g_lds = (_Float16*)(smem + GLDS_OFF);      // 4 KB
    float*    part  = (float*)(smem + PART_OFF);         // 2*64 floats

    const int tid   = threadIdx.x;
    const int lane  = tid & 31;
    const int wid   = tid >> 5;
    const int ct    = wid & 3;    // which 16-column tile of our 64 neurons
    const int kh    = wid >> 2;   // which K half (0..1023 / 1024..2047)
    const int wg    = blockIdx.x;
    const int gbase = wg * NPW;

    // preload this WG's weight fragments into LDS (1 KB per thread)
    {
        const char* src = (const char*)wf_g + (long)wg * WG_FRAG_BYTES;
        for (int off = tid * 16; off < WG_FRAG_BYTES; off += THREADS * 16)
            *(v4f*)(smem + off) = *(const v4f*)(src + off);
    }
    // zero the A-filler block (lanes that do not carry matrix row 0 read this)
    if (tid < 16) *(int*)(smem + ZBUF_OFF + tid * 4) = 0;

    // branch-free A addressing: row lanes (0,16) walk the g buffer, others pin
    // to the zero block with stride 0.
    const bool rowlane = (lane & 15) == 0;
    const int  aoff0   = rowlane ? (GLDS_OFF + kh * 2048 + ((lane >> 4) << 5))
                                 : ZBUF_OFF;
    const int  astep   = rowlane ? (KBLK * 2) : 0;       // 64 bytes per k-block

    // per-neuron state in registers of threads 0..63
    float e = 0.f, tm = 1.f, ri = 0.f, itg = 0.f, vcur = 0.f, gcur = 0.f;
    if (tid < NPW) {
        int gn = gbase + tid;
        e    = E_L[gn];
        tm   = tau_m[gn];
        ri   = R_I[gn];
        itg  = 1.f - 1.f / tau_g[gn];
        vcur = v0[gn];
        gcur = g0[gn];
        g_pub[gn] = (_Float16)gcur;    // publish g0 into ping buffer 0
    }
    grid_barrier(bar, tid, 1u);        // all g0 published before step 0 reads

    for (int t = 0; t < TSTEPS; ++t) {
        const _Float16* gin  = g_pub + (t & 1) * NN;
        _Float16*       gout = g_pub + ((t + 1) & 1) * NN;

        // broadcast g into LDS: 2048 halves, one b128 per thread
        *(v8h*)(g_lds + tid * 8) = *(const v8h*)(gin + tid * 8);
        __syncthreads();

        // ---- GEMV slice via WMMA: this wave covers cols [ct*16,+16), K half kh ----
        // 4 independent accumulators break the D->C serial chain in the matrix pipe.
        v8f acc0 = {}, acc1 = {}, acc2 = {}, acc3 = {};
        int ap = aoff0;
        int bp = (ct * NKB + kh * 32) * FRAG_BYTES + lane * 16;
#pragma unroll
        for (int kb = 0; kb < 8; ++kb) {
#pragma unroll
            for (int u = 0; u < 4; ++u) {
                v8h alo = *(const v8h*)(smem + ap);
                v8h ahi = *(const v8h*)(smem + ap + 16);
                v8h blo = *(const v8h*)(smem + bp);
                v8h bhi = *(const v8h*)(smem + bp + 512);
                v16h a, b;
#pragma unroll
                for (int i = 0; i < 8; ++i) {
                    a[i] = alo[i]; a[i + 8] = ahi[i];
                    b[i] = blo[i]; b[i + 8] = bhi[i];
                }
                if (u == 0)
                    acc0 = __builtin_amdgcn_wmma_f32_16x16x32_f16(
                               false, a, false, b, (short)0, acc0, false, false);
                else if (u == 1)
                    acc1 = __builtin_amdgcn_wmma_f32_16x16x32_f16(
                               false, a, false, b, (short)0, acc1, false, false);
                else if (u == 2)
                    acc2 = __builtin_amdgcn_wmma_f32_16x16x32_f16(
                               false, a, false, b, (short)0, acc2, false, false);
                else
                    acc3 = __builtin_amdgcn_wmma_f32_16x16x32_f16(
                               false, a, false, b, (short)0, acc3, false, false);
                ap += astep;
                bp += FRAG_BYTES;
            }
        }
        // reduce C rows (rows 1..15 are zero; sum is layout-robust)
        float s = 0.f;
#pragma unroll
        for (int i = 0; i < 8; ++i)
            s += (acc0[i] + acc1[i]) + (acc2[i] + acc3[i]);
        s += __shfl_xor(s, 16, 32);
        if (lane < 16) part[kh * 64 + ct * 16 + lane] = s;
        __syncthreads();

        // ---- neuron update (one thread per owned neuron) ----
        if (tid < NPW) {
            int gn   = gbase + tid;
            float x  = __builtin_nontemporal_load(x_in + (long)t * NN + gn);
            float I  = part[tid] + part[64 + tid] + 0.9f * x;
            float vn = vcur + (e - vcur + I * ri) / tm;
            float soft = 1.f / (1.f + __expf(30.f - vn));
            bool sp  = vn >= 30.f;
            vcur = sp ? e : vn;
            gcur = sp ? 1.f : gcur * itg;
            __builtin_nontemporal_store(soft, out + (long)t * NN + gn);
            gout[gn] = (_Float16)gcur;
        }
        grid_barrier(bar, tid, (unsigned)(t + 2));
    }
}

extern "C" void kernel_launch(void* const* d_in, const int* in_sizes, int n_in,
                              void* d_out, int out_size, void* d_ws, size_t ws_size,
                              hipStream_t stream) {
    const float* x_in  = (const float*)d_in[0];
    const float* w     = (const float*)d_in[1];
    const float* E_L   = (const float*)d_in[2];
    const float* tau_m = (const float*)d_in[3];
    const float* R_I   = (const float*)d_in[4];
    const float* tau_g = (const float*)d_in[5];
    const float* v0    = (const float*)d_in[6];
    const float* g0    = (const float*)d_in[7];
    float* out = (float*)d_out;

    char* ws = (char*)d_ws;
    unsigned* bar    = (unsigned*)ws;            // [0]=arrive count, [16]=epoch flag
    _Float16* g_pub  = (_Float16*)(ws + 256);    // ping-pong 2*2048 f16
    _Float16* wf     = (_Float16*)(ws + 16384);  // 8 MB fragmented f16 weights

    hipMemsetAsync(ws, 0, 256, stream);          // reset barrier state every launch

    lif_prep<<<(NN * NN / 16) / THREADS, THREADS, 0, stream>>>(w, wf);

    (void)hipFuncSetAttribute((const void*)lif_persist,
                              hipFuncAttributeMaxDynamicSharedMemorySize, SMEM_BYTES);
    lif_persist<<<NWG, THREADS, SMEM_BYTES, stream>>>(
        x_in, E_L, tau_m, R_I, tau_g, v0, g0, out, bar, g_pub, wf);
}
